// Gemma2Attention_32229434589677
// MI455X (gfx1250) — compile-verified, matrix operand order checked by reading, not run on hardware
//
#include <hip/hip_runtime.h>
#include <hip/hip_bf16.h>

// ---------------------------------------------------------------------------
// Gemma2 attention layer for MI455X (gfx1250), bf16 WMMA everywhere.
// S=4096, HID=3584, H=16, KV=8, D=256, window=1024, softcap=50, scale=1/16.
// ---------------------------------------------------------------------------

#define S_LEN   4096
#define HID     3584
#define NH      16
#define NKV     8
#define HD      256
#define QKVW    8192            // 16*256 Q + 8*256 K + 8*256 V per token row
#define VOFF    ((NH + NKV) * HD)   // 6144: V slot offset within a QKV row
#define WIN     1024
#define SOFTCAP 50.0f
#define INV_CAP 0.02f
#define QSCALE  0.0625f         // 256^-0.5
#define LOG_ROPE_BASE 9.210340371976184f  // ln(10000)

#if __has_builtin(__builtin_amdgcn_tensor_load_to_lds)
#define HAVE_TDM 1
#else
#define HAVE_TDM 0
#endif

typedef __attribute__((ext_vector_type(16))) __bf16 v16bf;
typedef __attribute__((ext_vector_type(8)))  float  v8f;

__device__ __forceinline__ v8f vzero8() {
  v8f z = {0.f,0.f,0.f,0.f,0.f,0.f,0.f,0.f};
  return z;
}

__device__ __forceinline__ float fast_tanh(float x) {
#if __has_builtin(__builtin_amdgcn_tanhf)
  return __builtin_amdgcn_tanhf(x);          // V_TANH_F32 (CDNA5 trans op)
#else
  return tanhf(x);
#endif
}

__device__ __forceinline__ v8f wmma_bf16(v16bf a, v16bf b, v8f c) {
  // D = A(16x32 bf16) * B(32x16 bf16) + C(16x16 f32)
  return __builtin_amdgcn_wmma_f32_16x16x32_bf16(
      /*neg_a=*/false, a, /*neg_b=*/false, b,
      /*c_mod=*/(short)0, c, /*reuse_a=*/false, /*reuse_b=*/false);
}

// K index held in VGPR r of a 16-bit A/B operand (doc: 16-bit A 16x32 layout).
__device__ __forceinline__ int kbase_of(int r, int half) {
  return ((r < 4) ? (2 * r) : (16 + 2 * (r - 4))) + 8 * half;
}

// Fragment where K is contiguous in memory: element (idx, k) at base[idx*ld + k].
__device__ __forceinline__ v16bf load_frag_kcontig(const __bf16* base, int ld) {
  int lane = threadIdx.x & 31;
  int half = lane >> 4, idx = lane & 15;
  union { unsigned int u[8]; v16bf v; } f;
  const __bf16* p = base + (long)idx * ld;
#pragma unroll
  for (int r = 0; r < 8; ++r) {
    int k = kbase_of(r, half);
    f.u[r] = *(const unsigned int*)(p + k);   // two bf16 (K=k,k+1) per dword
  }
  return f.v;
}

// --- CDNA5 async global->LDS copy (ASYNCcnt path, CDNA5 ISA) ----
__device__ __forceinline__ void async_g2l_b128(const __bf16* gsrc, __bf16* ldst) {
  unsigned lds_off = (unsigned)(uintptr_t)ldst;   // LDS aperture: offset in [31:0]
  asm volatile("global_load_async_to_lds_b128 %0, %1, off"
               :
               : "v"(lds_off), "v"(gsrc)
               : "memory");
}
template <int N>
__device__ __forceinline__ void wait_async() {
  asm volatile("s_wait_asynccnt %0" :: "n"(N) : "memory");
}

#if HAVE_TDM
// --- CDNA5 Tensor Data Mover: 2-D bf16 tile DMA, D# per 08_async_tensor §8 --
typedef unsigned int u32x4 __attribute__((ext_vector_type(4)));
typedef int          i32x8 __attribute__((ext_vector_type(8)));
typedef int          i32x4 __attribute__((ext_vector_type(4)));

__device__ __forceinline__ void tdm_load_2d_bf16(const __bf16* gsrc, __bf16* ldst,
                                                 int tile_w, int tile_h,
                                                 long row_stride_elems) {
  unsigned long long ga = (unsigned long long)gsrc;
  unsigned lds = (unsigned)(uintptr_t)ldst;
  u32x4 g0;
  g0[0] = 1u;                                            // count=1, user mode
  g0[1] = lds;                                           // lds_addr (bytes)
  g0[2] = (unsigned)(ga & 0xFFFFFFFFu);                  // global_addr[31:0]
  g0[3] = (unsigned)((ga >> 32) & 0x01FFFFFFu) | (2u << 30);  // ga[56:32] | type=2
  i32x8 g1;
  g1[0] = 1 << 16;                                       // data_size=2B
  g1[1] = (int)(((unsigned)tile_w & 0xFFFFu) << 16);     // tensor_dim0[15:0]
  g1[2] = (int)((((unsigned)tile_w >> 16) & 0xFFFFu) |   // tensor_dim0[31:16]
                (((unsigned)tile_h & 0xFFFFu) << 16));   // tensor_dim1[15:0]
  g1[3] = (int)((((unsigned)tile_h >> 16) & 0xFFFFu) |   // tensor_dim1[31:16]
                (((unsigned)tile_w & 0xFFFFu) << 16));   // tile_dim0
  g1[4] = (int)((unsigned)tile_h & 0xFFFFu);             // tile_dim1 | tile_dim2=0
  g1[5] = (int)(row_stride_elems & 0xFFFFFFFFu);         // dim0_stride[31:0]
  g1[6] = (int)((row_stride_elems >> 32) & 0xFFFFu);     // dim0_stride[47:32]
  g1[7] = 0;
  i32x4 z4 = {0, 0, 0, 0};
  i32x8 z8 = {0, 0, 0, 0, 0, 0, 0, 0};
  // clang-23 (therock-10.0) 6-arg form: groups 0..3 + extra group + cpol
  __builtin_amdgcn_tensor_load_to_lds(g0, g1, z4, z4, z8, 0);
}
#endif

// ---------------------------------------------------------------------------
// fp32 -> bf16 bulk convert (n multiple of 1024)
// ---------------------------------------------------------------------------
__global__ __launch_bounds__(256)
void cvt_f32_bf16(const float* __restrict__ src, __bf16* __restrict__ dst, long n) {
  long i = ((long)blockIdx.x * 256 + threadIdx.x) * 4;
  if (i + 3 < n) {
    float4 v = *(const float4*)(src + i);
    union { __bf16 e[4]; uint2 u; } p;
    p.e[0] = (__bf16)v.x; p.e[1] = (__bf16)v.y;
    p.e[2] = (__bf16)v.z; p.e[3] = (__bf16)v.w;
    *(uint2*)(dst + i) = p.u;
  }
}

// ---------------------------------------------------------------------------
// Tiled GEMM: C[M,N] = A[M,K] * B[N,K]^T, bf16 in, f32 accumulate.
// Block 256 threads = 8 waves; tile 128x128x32; double-buffered staging
// via TDM (wave 0 issues tile DMA) or per-lane async copies.
// ---------------------------------------------------------------------------
template <bool OUT_BF16>
__global__ __launch_bounds__(256)
void gemm_nt_bf16(const __bf16* __restrict__ A, const __bf16* __restrict__ B,
                  void* __restrict__ Cout, int M, int N, int K) {
  constexpr int BM = 128, BN = 128, KT = 32;
  __shared__ __align__(16) __bf16 As[2][BM * KT];
  __shared__ __align__(16) __bf16 Bs[2][BN * KT];

  const int m0 = blockIdx.y * BM, n0 = blockIdx.x * BN;
  const int tid = threadIdx.x;
  const int wave = tid >> 5;
  const int lane = tid & 31, half = lane >> 4, idx = lane & 15;

  v8f acc[8];
#pragma unroll
  for (int i = 0; i < 8; ++i) acc[i] = vzero8();

  // stage(k0, buf)
  auto stage = [&](int k0, int b) {
#if HAVE_TDM
    if (wave == 0) {
      tdm_load_2d_bf16(A + (long)m0 * K + k0, &As[b][0], KT, BM, K);
      tdm_load_2d_bf16(B + (long)n0 * K + k0, &Bs[b][0], KT, BN, K);
    }
#else
#pragma unroll
    for (int i = 0; i < 2; ++i) {
      int v   = tid + i * 256;
      int row = v >> 2, seg = v & 3;
      async_g2l_b128(A + (long)(m0 + row) * K + k0 + seg * 8, &As[b][row * KT + seg * 8]);
      async_g2l_b128(B + (long)(n0 + row) * K + k0 + seg * 8, &Bs[b][row * KT + seg * 8]);
    }
#endif
  };

  int cur = 0;
  stage(0, 0);
  for (int k0 = 0; k0 < K; k0 += KT) {
    if (k0 + KT < K) {
      stage(k0 + KT, cur ^ 1);
#if HAVE_TDM
      if (wave == 0) __builtin_amdgcn_s_wait_tensorcnt(2);
#else
      wait_async<4>();
#endif
    } else {
#if HAVE_TDM
      if (wave == 0) __builtin_amdgcn_s_wait_tensorcnt(0);
#else
      wait_async<0>();
#endif
    }
    __syncthreads();

    v16bf af = load_frag_kcontig(&As[cur][wave * 16 * KT], KT);
#pragma unroll
    for (int nt = 0; nt < 8; ++nt) {
      v16bf bf = load_frag_kcontig(&Bs[cur][nt * 16 * KT], KT);
      acc[nt] = wmma_bf16(af, bf, acc[nt]);
    }
    __syncthreads();
    cur ^= 1;
  }

#pragma unroll
  for (int nt = 0; nt < 8; ++nt) {
#pragma unroll
    for (int r = 0; r < 8; ++r) {
      long row = m0 + wave * 16 + r + 8 * half;
      long col = n0 + nt * 16 + idx;
      float val = acc[nt][r];
      if (OUT_BF16) ((__bf16*)Cout)[row * N + col] = (__bf16)val;
      else          ((float*)Cout)[row * N + col]  = val;
    }
  }
}

// ---------------------------------------------------------------------------
// In-place interleaved-pair RoPE on QKV rows [S, 8192]; slots 0..15 = Q heads
// (also scaled by 1/16), slots 16..23 = K heads. V untouched.
// ---------------------------------------------------------------------------
__global__ __launch_bounds__(128)
void rope_inplace(__bf16* __restrict__ qkv) {
  const int s    = blockIdx.x;
  const int slot = blockIdx.y;          // 0..23
  const int t    = threadIdx.x;         // pair index 0..127
  long base = (long)s * QKVW + slot * HD;

  float freq = __expf(-((2.0f * t) / (float)HD) * LOG_ROPE_BASE);
  float ang  = (float)s * freq;
  float sn, cs;
  __sincosf(ang, &sn, &cs);

  float x0 = (float)qkv[base + 2 * t];
  float x1 = (float)qkv[base + 2 * t + 1];
  float y0 = x0 * cs - x1 * sn;
  float y1 = x1 * cs + x0 * sn;
  if (slot < NH) { y0 *= QSCALE; y1 *= QSCALE; }
  qkv[base + 2 * t]     = (__bf16)y0;
  qkv[base + 2 * t + 1] = (__bf16)y1;
}

// ---------------------------------------------------------------------------
// V transpose: Vt[kv][d][s] = QKV[s][VOFF + kv*256 + d]   (LDS 32x32 tiles)
// ---------------------------------------------------------------------------
__global__ __launch_bounds__(256)
void transpose_v(const __bf16* __restrict__ qkv, __bf16* __restrict__ Vt) {
  __shared__ __bf16 t[32][33];
  const int s0 = blockIdx.x * 32, d0 = blockIdx.y * 32, kv = blockIdx.z;
  const int tx = threadIdx.x & 31, ty = threadIdx.x >> 5;   // 32 x 8
#pragma unroll
  for (int r = 0; r < 4; ++r) {
    int row = ty + r * 8;
    t[row][tx] = qkv[(long)(s0 + row) * QKVW + VOFF + kv * HD + d0 + tx];
  }
  __syncthreads();
#pragma unroll
  for (int r = 0; r < 4; ++r) {
    int drow = ty + r * 8;
    Vt[((long)kv * HD + d0 + drow) * S_LEN + s0 + tx] = t[tx][drow];
  }
}

// ---------------------------------------------------------------------------
// Flash attention with causal + sliding-window band and tanh softcapping.
// grid = (S/128, H); block = 256 (8 waves); wave owns a 16-row query tile.
// Double-buffered K/V tile staging: TDM tile DMA (wave 0) or async copies.
// ---------------------------------------------------------------------------
__global__ __launch_bounds__(256)
void attn_swa(const __bf16* __restrict__ qkv, const __bf16* __restrict__ Vt,
              __bf16* __restrict__ O) {
  __shared__ __align__(16) __bf16 Ks[2][32 * HD];   // [key][d]   2x16 KB
  __shared__ __align__(16) __bf16 Vs[2][HD * 32];   // [d][key]   2x16 KB
  __shared__ __bf16 Pl[8][16 * 32];                 // per-wave P   8 KB

  const int h    = blockIdx.y;
  const int kvh  = h >> 1;              // GQA repeat = 2
  const int tid  = threadIdx.x;
  const int wave = tid >> 5;
  const int lane = tid & 31, half = lane >> 4, idx = lane & 15;
  const int qblk = blockIdx.x * 128;
  const int qlo  = qblk + wave * 16;

  // Q fragments: 16 rows x 256 dims, 8 k-chunks of 32.
  const __bf16* Qb = qkv + (long)qlo * QKVW + h * HD;
  v16bf qf[8];
#pragma unroll
  for (int c = 0; c < 8; ++c) qf[c] = load_frag_kcontig(Qb + c * 32, QKVW);

  v8f o[16];
#pragma unroll
  for (int i = 0; i < 16; ++i) o[i] = vzero8();
  float m[8], l[8];
#pragma unroll
  for (int r = 0; r < 8; ++r) { m[r] = -1e30f; l[r] = 0.f; }

  int jmin = qblk - (WIN - 1); if (jmin < 0) jmin = 0;
  const int kt0  = jmin & ~31;
  const int kend = qblk + 128;

  const __bf16* KgBase = qkv + NH * HD + (long)kvh * HD;   // + kt*QKVW
  const __bf16* VgBase = Vt + (long)kvh * HD * S_LEN;      // + kt

  auto stage = [&](int kt, int b) {
#if HAVE_TDM
    if (wave == 0) {
      tdm_load_2d_bf16(KgBase + (long)kt * QKVW, &Ks[b][0], HD, 32, QKVW);
      tdm_load_2d_bf16(VgBase + kt,              &Vs[b][0], 32, HD, S_LEN);
    }
#else
#pragma unroll
    for (int i = 0; i < 4; ++i) {
      int v = tid + i * 256;
      int krow = v >> 5, kseg = v & 31;
      async_g2l_b128(KgBase + (long)(kt + krow) * QKVW + kseg * 8,
                     &Ks[b][krow * HD + kseg * 8]);
      int vrow = v >> 2, vseg = v & 3;
      async_g2l_b128(VgBase + (long)vrow * S_LEN + kt + vseg * 8,
                     &Vs[b][vrow * 32 + vseg * 8]);
    }
#endif
  };

  int cur = 0;
  stage(kt0, 0);
  for (int kt = kt0; kt < kend; kt += 32) {
    if (kt + 32 < kend) {
      stage(kt + 32, cur ^ 1);
#if HAVE_TDM
      if (wave == 0) __builtin_amdgcn_s_wait_tensorcnt(2);
#else
      wait_async<8>();
#endif
    } else {
#if HAVE_TDM
      if (wave == 0) __builtin_amdgcn_s_wait_tensorcnt(0);
#else
      wait_async<0>();
#endif
    }
    __syncthreads();

    // ---- scores: two 16x16 C tiles (keys kt..+15, kt+16..+31) ------------
    v8f s0 = vzero8(), s1 = vzero8();
#pragma unroll
    for (int c = 0; c < 8; ++c) {
      v16bf k0 = load_frag_kcontig(&Ks[cur][c * 32], HD);
      v16bf k1 = load_frag_kcontig(&Ks[cur][16 * HD + c * 32], HD);
      s0 = wmma_bf16(qf[c], k0, s0);
      s1 = wmma_bf16(qf[c], k1, s1);
    }

    float a_r[8], p0[8], p1[8];
    bool changed = false;
#pragma unroll
    for (int r = 0; r < 8; ++r) {
      int i  = qlo + r + 8 * half;      // query row of this accumulator slot
      int j0 = kt + idx, j1 = j0 + 16;  // key column (lane)
      float v0 = SOFTCAP * fast_tanh(s0[r] * INV_CAP);
      float v1 = SOFTCAP * fast_tanh(s1[r] * INV_CAP);
      bool ok0 = (j0 <= i) && (j0 >= i - (WIN - 1));
      bool ok1 = (j1 <= i) && (j1 >= i - (WIN - 1));
      float mx = fmaxf(ok0 ? v0 : -1e30f, ok1 ? v1 : -1e30f);
      mx = fmaxf(mx, __shfl_xor(mx, 1));
      mx = fmaxf(mx, __shfl_xor(mx, 2));
      mx = fmaxf(mx, __shfl_xor(mx, 4));
      mx = fmaxf(mx, __shfl_xor(mx, 8));       // row max within 16-lane half
      float mn = fmaxf(m[r], mx);
      float a  = __expf(m[r] - mn);
      float e0 = ok0 ? __expf(v0 - mn) : 0.f;
      float e1 = ok1 ? __expf(v1 - mn) : 0.f;
      float rs = e0 + e1;
      rs += __shfl_xor(rs, 1);
      rs += __shfl_xor(rs, 2);
      rs += __shfl_xor(rs, 4);
      rs += __shfl_xor(rs, 8);
      l[r] = l[r] * a + rs;
      m[r] = mn;
      a_r[r] = a; p0[r] = e0; p1[r] = e1;
      changed |= (a != 1.0f);
    }
    if (__any(changed)) {               // skip rescale when no row max moved
#pragma unroll
      for (int nt = 0; nt < 16; ++nt)
#pragma unroll
        for (int r = 0; r < 8; ++r) o[nt][r] *= a_r[r];
    }

    // ---- stage P (16x32 bf16) in LDS: C-tile layout -> A-operand ---------
    __bf16* pw = &Pl[wave][0];
#pragma unroll
    for (int r = 0; r < 8; ++r) {
      int row = r + 8 * half;
      pw[row * 32 + idx]      = (__bf16)p0[r];
      pw[row * 32 + idx + 16] = (__bf16)p1[r];
    }
    asm volatile("s_wait_dscnt 0x0" ::: "memory");   // CDNA5 split DS counter
    v16bf pf = load_frag_kcontig(pw, 32);

    // ---- O += P * V ------------------------------------------------------
#pragma unroll
    for (int nt = 0; nt < 16; ++nt) {
      v16bf vf = load_frag_kcontig(&Vs[cur][nt * 16 * 32], 32);
      o[nt] = wmma_bf16(pf, vf, o[nt]);
    }
    __syncthreads();                     // all reads done before next stage
    cur ^= 1;
  }

  // normalize and store O[s][h*256 + d] as bf16
  float inv[8];
#pragma unroll
  for (int r = 0; r < 8; ++r) inv[r] = (l[r] > 0.f) ? (1.f / l[r]) : 0.f;
#pragma unroll
  for (int nt = 0; nt < 16; ++nt)
#pragma unroll
    for (int r = 0; r < 8; ++r) {
      long row = qlo + r + 8 * half;
      long col = (long)h * HD + nt * 16 + idx;
      O[row * (NH * HD) + col] = (__bf16)(o[nt][r] * inv[r]);
    }
}

// ---------------------------------------------------------------------------
// launch
// ---------------------------------------------------------------------------
extern "C" void kernel_launch(void* const* d_in, const int* in_sizes, int n_in,
                              void* d_out, int out_size, void* d_ws, size_t ws_size,
                              hipStream_t stream) {
  const float* x  = (const float*)d_in[0];   // [1,4096,3584]
  // d_in[1] = mask (unused: band computed analytically)
  const float* Wq = (const float*)d_in[2];   // [4096,3584]
  const float* Wk = (const float*)d_in[3];   // [2048,3584]
  const float* Wv = (const float*)d_in[4];   // [2048,3584]
  const float* Wo = (const float*)d_in[5];   // [3584,4096]
  // d_in[6] = input_pos (arange; positions derived from index)
  float* out = (float*)d_out;                // [4096,3584]

  char* ws = (char*)d_ws;
  size_t off = 0;
  __bf16* Xbf  = (__bf16*)(ws + off); off += (size_t)S_LEN * HID * 2;    //  28 MB
  __bf16* Wqkv = (__bf16*)(ws + off); off += (size_t)QKVW  * HID * 2;    //  56 MB
  __bf16* Wob  = (__bf16*)(ws + off); off += (size_t)HID * (NH*HD) * 2;  //  28 MB
  __bf16* QKV  = (__bf16*)(ws + off); off += (size_t)S_LEN * QKVW * 2;   //  64 MB
  __bf16* Vt   = (__bf16*)(ws + off); off += (size_t)NKV * HD * S_LEN*2; //  16 MB
  __bf16* Obf  = (__bf16*)(ws + off); off += (size_t)S_LEN * (NH*HD)*2;  //  32 MB
  (void)off; (void)ws_size; (void)in_sizes; (void)n_in; (void)out_size;

  const long nX  = (long)S_LEN * HID;
  const long nWq = (long)NH  * HD * HID;
  const long nWk = (long)NKV * HD * HID;
  const long nWo = (long)HID * NH * HD;

  cvt_f32_bf16<<<dim3(nX  / 1024), 256, 0, stream>>>(x,  Xbf, nX);
  cvt_f32_bf16<<<dim3(nWq / 1024), 256, 0, stream>>>(Wq, Wqkv, nWq);
  cvt_f32_bf16<<<dim3(nWk / 1024), 256, 0, stream>>>(Wk, Wqkv + (size_t)NH * HD * HID, nWk);
  cvt_f32_bf16<<<dim3(nWk / 1024), 256, 0, stream>>>(Wv, Wqkv + (size_t)(NH + NKV) * HD * HID, nWk);
  cvt_f32_bf16<<<dim3(nWo / 1024), 256, 0, stream>>>(Wo, Wob, nWo);

  // QKV = X * Wqkv^T   [4096 x 8192]
  gemm_nt_bf16<true><<<dim3(QKVW / 128, S_LEN / 128), 256, 0, stream>>>(
      Xbf, Wqkv, QKV, S_LEN, QKVW, HID);

  // RoPE on Q (scaled) + K, in place
  rope_inplace<<<dim3(S_LEN, NH + NKV), 128, 0, stream>>>(QKV);

  // V transpose for packed PV operands
  transpose_v<<<dim3(S_LEN / 32, HD / 32, NKV), 256, 0, stream>>>(QKV, Vt);

  // flash attention with sliding-window band
  attn_swa<<<dim3(S_LEN / 128, NH), 256, 0, stream>>>(QKV, Vt, Obf);

  // out = O * Wo^T   [4096 x 3584] fp32
  gemm_nt_bf16<false><<<dim3(HID / 128, S_LEN / 128), 256, 0, stream>>>(
      Obf, Wob, out, S_LEN, HID, NH * HD);
}
